// LSTM_51616916963993
// MI455X (gfx1250) — matrix operand order repaired and checked by reading, not compiled, not run on hardware
//
#include <hip/hip_runtime.h>

// ---------------------------------------------------------------------------
// LSTM (S=512, B=64, I=256, H=512) for gfx1250 (MI455X), wave32 + WMMA f16.
//
// Phase 0: convert weights/x/h to f16, combine biases, init c state.
// Phase 1: P[g][s*B+b][n] = x[s,b,:] @ Wg_i^T + (b_gi + b_gh)   (big GEMM)
// Phase 2: 512 sequential step kernels: TDM-stage h tile to LDS, then
//          acc = P + h @ Wg_h^T via WMMA (A from LDS), fused cell update,
//          double-buffered f16 h.
// ---------------------------------------------------------------------------

typedef __attribute__((ext_vector_type(16))) _Float16     v16h;
typedef __attribute__((ext_vector_type(8)))  _Float16     v8h;
typedef __attribute__((ext_vector_type(8)))  float        v8f;
typedef __attribute__((ext_vector_type(4)))  unsigned int v4u;
typedef __attribute__((ext_vector_type(8)))  int          v8i_t;
typedef __attribute__((ext_vector_type(4)))  int          v4i_t;

#define S_LEN 512
#define B_DIM 64
#define I_DIM 256
#define H_DIM 512

#define HAVE_TDM (__has_builtin(__builtin_amdgcn_tensor_load_to_lds) && \
                  __has_builtin(__builtin_amdgcn_s_wait_tensorcnt))

union V16U { v16h v; v8h h[2]; };

// A fragment: 16x32 f16 tile, row-major source, tile base = src + m0*lda + k0.
// Lane L (0-15): M=L, holds K = {0..7, 16..23}; lane L+16: M=L, K = {8..15, 24..31}.
__device__ __forceinline__ v16h load_a_frag(const _Float16* tile, int lda, int lane) {
    int m  = lane & 15;
    int kb = (lane & 16) ? 8 : 0;
    const _Float16* p = tile + m * lda + kb;
    V16U r;
    r.h[0] = *(const v8h*)(p);
    r.h[1] = *(const v8h*)(p + 16);
    return r.v;
}

// B fragment: 32x16 f16 (KxN). Column n of B == row n of the row-major weight
// matrix W[n][k]. Lane n (0-15): K = {0..7, 16..23} of column n; lane n+16: K = {8..15, 24..31}.
__device__ __forceinline__ v16h load_b_frag(const _Float16* tile, int ldw, int lane) {
    int n  = lane & 15;
    int kb = (lane & 16) ? 8 : 0;
    const _Float16* p = tile + n * ldw + kb;
    V16U r;
    r.h[0] = *(const v8h*)(p);
    r.h[1] = *(const v8h*)(p + 16);
    return r.v;
}

__device__ __forceinline__ float sigmoidf_fast(float x) {
    return 1.0f / (1.0f + __expf(-x));
}

// ---------------------------------------------------------------------------
// Phase 0 helpers
// ---------------------------------------------------------------------------
__global__ void cvt_f32_to_f16_kernel(const float* __restrict__ src,
                                      _Float16* __restrict__ dst, int n) {
    int i = blockIdx.x * blockDim.x + threadIdx.x;
    if (i < n) dst[i] = (_Float16)src[i];
}

__global__ void bias_sum_kernel(const float* __restrict__ bi,
                                const float* __restrict__ bh,
                                float* __restrict__ bias, int n) {
    int i = blockIdx.x * blockDim.x + threadIdx.x;
    if (i < n) bias[i] = bi[i] + bh[i];
}

__global__ void copy_f32_kernel(const float* __restrict__ src,
                                float* __restrict__ dst, int n) {
    int i = blockIdx.x * blockDim.x + threadIdx.x;
    if (i < n) dst[i] = src[i];
}

// ---------------------------------------------------------------------------
// Phase 1: input projections for all gates / all timesteps.
// Grid: (S*B/16, H/16/4). Block = 128 threads (4 waves); each wave owns one
// (m-tile, n-tile). A fragments (K=256 -> 8 frags) are loaded ONCE and held
// in registers across all 4 gates.
// ---------------------------------------------------------------------------
__global__ void __launch_bounds__(128)
lstm_proj_kernel(const _Float16* __restrict__ x16,   // [S*B][I]
                 const _Float16* __restrict__ Wi16,  // [4][H][I]
                 const float*    __restrict__ bias,  // [4][H]
                 _Float16*       __restrict__ P)     // [4][S*B][H]
{
    const int lane = threadIdx.x & 31;
    const int wave = threadIdx.x >> 5;
    const int m0 = blockIdx.x * 16;
    const int nt = blockIdx.y * 4 + wave;
    const int n0 = nt * 16;
    const int n  = n0 + (lane & 15);
    const int mr = (lane & 16) ? 8 : 0;

    const _Float16* Arow = x16 + (size_t)m0 * I_DIM;

    v16h aF[I_DIM / 32];
    #pragma unroll
    for (int kk = 0; kk < I_DIM / 32; ++kk)
        aF[kk] = load_a_frag(Arow + kk * 32, I_DIM, lane);

    #pragma unroll
    for (int g = 0; g < 4; ++g) {
        float bv = bias[g * H_DIM + n];
        v8f acc;
        #pragma unroll
        for (int j = 0; j < 8; ++j) acc[j] = bv;

        const _Float16* Wg = Wi16 + (size_t)g * H_DIM * I_DIM + (size_t)n0 * I_DIM;
        #pragma unroll
        for (int kk = 0; kk < I_DIM / 32; ++kk) {
            v16h b = load_b_frag(Wg + kk * 32, I_DIM, lane);
            acc = __builtin_amdgcn_wmma_f32_16x16x32_f16(
                false, aF[kk], false, b, (short)0, acc, false, false);
        }

        _Float16* Pt = P + ((size_t)g * S_LEN * B_DIM + m0) * H_DIM + n;
        #pragma unroll
        for (int j = 0; j < 8; ++j)
            Pt[(size_t)(mr + j) * H_DIM] = (_Float16)acc[j];
    }
}

// ---------------------------------------------------------------------------
// Phase 2: one timestep. Grid (B/16, H/16/4), block = 128 (4 waves).
// The 16x512 f16 h-tile (16 KB, contiguous) shared by all 4 waves is staged
// into LDS by one TDM tensor_load_to_lds (ASYNC/TENSOR path), then each wave
// computes its 16x16 (b,n) tile for all 4 gates with A fragments from LDS,
// followed by the fused LSTM cell update on the accumulator layout.
// ---------------------------------------------------------------------------
__global__ void __launch_bounds__(128)
lstm_step_kernel(const _Float16* __restrict__ P,      // [4][S*B][H]
                 const _Float16* __restrict__ Wh16,   // [4][H][H]
                 const _Float16* __restrict__ h_in,   // [B][H] f16
                 _Float16*       __restrict__ h_out,  // [B][H] f16
                 float*          __restrict__ c_state,// [B][H] f32
                 float*          __restrict__ out,    // d_out
                 int s)
{
    const int lane = threadIdx.x & 31;
    const int wave = threadIdx.x >> 5;
    const int b0 = blockIdx.x * 16;
    const int nt = blockIdx.y * 4 + wave;
    const int n0 = nt * 16;
    const int n  = n0 + (lane & 15);
    const int mr = (lane & 16) ? 8 : 0;

    __shared__ __align__(16) _Float16 hTile[16 * H_DIM];  // 16 KB

#if HAVE_TDM
    if (wave == 0) {
        // ---- Tensor DMA: contiguous 1-D tile of 8192 f16 into LDS ----
        unsigned long long gaddr =
            (unsigned long long)(const void*)(h_in + (size_t)b0 * H_DIM);
        unsigned lds_off = (unsigned)(unsigned long long)(void*)hTile;

        v4u g0;
        g0[0] = 1u;                                          // count = 1 descriptor
        g0[1] = lds_off;                                     // lds_addr (bytes)
        g0[2] = (unsigned)(gaddr & 0xFFFFFFFFull);           // global_addr[31:0]
        g0[3] = (unsigned)((gaddr >> 32) & 0x01FFFFFFull)    // global_addr[56:32]
              | (2u << 30);                                  // type = 2 (image)

        v8i_t g1 = (v8i_t)0;
        g1[0] = (int)(1u << 16);        // wg_mask=0, data_size=1 (2 bytes)
        g1[1] = (int)(8192u << 16);     // tensor_dim0 = 8192 (low 16 bits)
        g1[2] = (int)(1u << 16);        // tensor_dim0 hi=0 ; tensor_dim1 = 1
        g1[3] = (int)(8192u << 16);     // tensor_dim1 hi=0 ; tile_dim0 = 8192
        g1[4] = 0;                      // tile_dim1 = 0 (unused), tile_dim2 = 0
        g1[5] = (int)8192u;             // tensor_dim0_stride = 8192
        g1[6] = 0;
        g1[7] = 0;

        v4i_t gz = (v4i_t)0;
#if __clang_major__ >= 23
        v8i_t gz8 = (v8i_t)0;
        __builtin_amdgcn_tensor_load_to_lds(g0, g1, gz, gz, gz8, 0);
#else
        __builtin_amdgcn_tensor_load_to_lds(g0, g1, gz, gz, 0);
#endif
        __builtin_amdgcn_s_wait_tensorcnt(0);
    }
    __syncthreads();
#else
    {   // Fallback: cooperative 16 KB copy through VGPRs (still uses LDS).
        const v8h* src = (const v8h*)(h_in + (size_t)b0 * H_DIM);
        v8h* dst = (v8h*)hTile;
        for (int i = threadIdx.x; i < 16 * H_DIM / 8; i += 128)
            dst[i] = src[i];
    }
    __syncthreads();
#endif

    // Init accumulators with the precomputed input projections (+biases).
    v8f acc[4];
    #pragma unroll
    for (int g = 0; g < 4; ++g) {
        const _Float16* Pt =
            P + ((size_t)g * S_LEN * B_DIM + (size_t)s * B_DIM + b0) * H_DIM + n;
        #pragma unroll
        for (int j = 0; j < 8; ++j)
            acc[g][j] = (float)Pt[(size_t)(mr + j) * H_DIM];
    }

    // Recurrent GEMM: K = H = 512. A from LDS; all 4 gate B fragments are
    // loaded into distinct registers first so the loads clause & overlap.
    const _Float16* W0 = Wh16 + ((size_t)0 * H_DIM + n0) * H_DIM;
    const _Float16* W1 = Wh16 + ((size_t)1 * H_DIM + n0) * H_DIM;
    const _Float16* W2 = Wh16 + ((size_t)2 * H_DIM + n0) * H_DIM;
    const _Float16* W3 = Wh16 + ((size_t)3 * H_DIM + n0) * H_DIM;

    #pragma unroll 4
    for (int k0 = 0; k0 < H_DIM; k0 += 32) {
        v16h a  = load_a_frag(hTile + k0, H_DIM, lane);
        v16h bf0 = load_b_frag(W0 + k0, H_DIM, lane);
        v16h bf1 = load_b_frag(W1 + k0, H_DIM, lane);
        v16h bf2 = load_b_frag(W2 + k0, H_DIM, lane);
        v16h bf3 = load_b_frag(W3 + k0, H_DIM, lane);
        acc[0] = __builtin_amdgcn_wmma_f32_16x16x32_f16(
            false, a, false, bf0, (short)0, acc[0], false, false);
        acc[1] = __builtin_amdgcn_wmma_f32_16x16x32_f16(
            false, a, false, bf1, (short)0, acc[1], false, false);
        acc[2] = __builtin_amdgcn_wmma_f32_16x16x32_f16(
            false, a, false, bf2, (short)0, acc[2], false, false);
        acc[3] = __builtin_amdgcn_wmma_f32_16x16x32_f16(
            false, a, false, bf3, (short)0, acc[3], false, false);
    }

    // Fused gate nonlinearities + cell update, directly on the C/D layout.
    const size_t seq_off = (size_t)s * B_DIM * H_DIM;
    const size_t fin_off = (size_t)S_LEN * B_DIM * H_DIM;
    #pragma unroll
    for (int j = 0; j < 8; ++j) {
        int m = b0 + mr + j;
        size_t idx = (size_t)m * H_DIM + n;
        float f_t  = sigmoidf_fast(acc[0][j]);
        float i_t  = sigmoidf_fast(acc[1][j]);
        float ct_t = tanhf(acc[2][j]);
        float o_t  = sigmoidf_fast(acc[3][j]);
        float c    = c_state[idx] * f_t + i_t * ct_t;
        c_state[idx] = c;
        float h = o_t * tanhf(c);
        out[seq_off + idx] = h;          // h_seq[s]
        h_out[idx] = (_Float16)h;        // next-step recurrent input
        if (s == S_LEN - 1) {
            out[fin_off + idx] = h;                          // final h
            out[fin_off + (size_t)B_DIM * H_DIM + idx] = c;  // final c
        }
    }
}

// ---------------------------------------------------------------------------
// Host launcher
// ---------------------------------------------------------------------------
extern "C" void kernel_launch(void* const* d_in, const int* in_sizes, int n_in,
                              void* d_out, int out_size, void* d_ws, size_t ws_size,
                              hipStream_t stream) {
    (void)in_sizes; (void)n_in; (void)out_size; (void)ws_size;

    const float* x = (const float*)d_in[0];
    const float* h0 = (const float*)d_in[1];
    const float* c0 = (const float*)d_in[2];
    // Gate order: 0=f, 1=i, 2=c(candidate), 3=o
    const float* Wi[4] = { (const float*)d_in[3],  (const float*)d_in[7],
                           (const float*)d_in[11], (const float*)d_in[15] };
    const float* bi[4] = { (const float*)d_in[4],  (const float*)d_in[8],
                           (const float*)d_in[12], (const float*)d_in[16] };
    const float* Wh[4] = { (const float*)d_in[5],  (const float*)d_in[9],
                           (const float*)d_in[13], (const float*)d_in[17] };
    const float* bh[4] = { (const float*)d_in[6],  (const float*)d_in[10],
                           (const float*)d_in[14], (const float*)d_in[18] };
    float* out = (float*)d_out;

    // Workspace layout (bytes)
    char* ws = (char*)d_ws;
    const size_t P_bytes    = (size_t)4 * S_LEN * B_DIM * H_DIM * sizeof(_Float16); // 128 MB
    const size_t Wi16_bytes = (size_t)4 * H_DIM * I_DIM * sizeof(_Float16);
    const size_t Wh16_bytes = (size_t)4 * H_DIM * H_DIM * sizeof(_Float16);
    const size_t x16_bytes  = (size_t)S_LEN * B_DIM * I_DIM * sizeof(_Float16);
    const size_t h16_bytes  = (size_t)2 * B_DIM * H_DIM * sizeof(_Float16); // double buffer
    const size_t c_bytes    = (size_t)B_DIM * H_DIM * sizeof(float);

    _Float16* P    = (_Float16*)(ws);
    _Float16* Wi16 = (_Float16*)(ws + P_bytes);
    _Float16* Wh16 = (_Float16*)(ws + P_bytes + Wi16_bytes);
    _Float16* x16  = (_Float16*)(ws + P_bytes + Wi16_bytes + Wh16_bytes);
    _Float16* h16  = (_Float16*)(ws + P_bytes + Wi16_bytes + Wh16_bytes + x16_bytes);
    float*    cst  = (float*)   (ws + P_bytes + Wi16_bytes + Wh16_bytes + x16_bytes + h16_bytes);
    float*    bias = (float*)   (ws + P_bytes + Wi16_bytes + Wh16_bytes + x16_bytes + h16_bytes + c_bytes);

    const int TPB = 256;
    auto blocks = [](int n, int tpb) { return (n + tpb - 1) / tpb; };

    // Phase 0: conversions / init
    {
        int n = S_LEN * B_DIM * I_DIM;
        cvt_f32_to_f16_kernel<<<blocks(n, TPB), TPB, 0, stream>>>(x, x16, n);
    }
    for (int g = 0; g < 4; ++g) {
        int ni = H_DIM * I_DIM;
        cvt_f32_to_f16_kernel<<<blocks(ni, TPB), TPB, 0, stream>>>(
            Wi[g], Wi16 + (size_t)g * H_DIM * I_DIM, ni);
        int nh = H_DIM * H_DIM;
        cvt_f32_to_f16_kernel<<<blocks(nh, TPB), TPB, 0, stream>>>(
            Wh[g], Wh16 + (size_t)g * H_DIM * H_DIM, nh);
        bias_sum_kernel<<<blocks(H_DIM, TPB), TPB, 0, stream>>>(
            bi[g], bh[g], bias + g * H_DIM, H_DIM);
    }
    {
        int n = B_DIM * H_DIM;
        cvt_f32_to_f16_kernel<<<blocks(n, TPB), TPB, 0, stream>>>(h0, h16, n);
        copy_f32_kernel<<<blocks(n, TPB), TPB, 0, stream>>>(c0, cst, n);
    }

    // Phase 1: big input-projection GEMM (parallel across all timesteps)
    {
        dim3 grid(S_LEN * B_DIM / 16, H_DIM / 16 / 4);
        lstm_proj_kernel<<<grid, 128, 0, stream>>>(x16, Wi16, bias, P);
    }

    // Phase 2: sequential recurrence, double-buffered f16 h
    {
        dim3 grid(B_DIM / 16, H_DIM / 16 / 4);
        const size_t hbuf = (size_t)B_DIM * H_DIM;
        for (int s = 0; s < S_LEN; ++s) {
            const _Float16* hin  = h16 + (size_t)(s & 1) * hbuf;
            _Float16*       hout = h16 + (size_t)((s + 1) & 1) * hbuf;
            lstm_step_kernel<<<grid, 128, 0, stream>>>(
                P, Wh16, hin, hout, cst, out, s);
        }
    }
}
